// PairformerBlock_53377853555086
// MI455X (gfx1250) — matrix operand order repaired and checked by reading, not compiled, not run on hardware
//
#include <hip/hip_runtime.h>
#include <hip/hip_bf16.h>

typedef _Float16 h16;
typedef __attribute__((ext_vector_type(16))) _Float16 v16h;
typedef __attribute__((ext_vector_type(8)))  _Float16 v8h;
typedef __attribute__((ext_vector_type(8)))  float    v8f;

#define DEV __device__ __forceinline__

DEV v8f wmma_f16(v16h a, v16h b, v8f c) {
  return __builtin_amdgcn_wmma_f32_16x16x32_f16(false, a, false, b, (short)0, c, false, false);
}

// A fragment (V_WMMA_F32_16X16X32_F16): lane m = lane&15, half = lane>>4.
// elems 0..15 <- K = 16*half + 0..15
DEV v16h load_a_frag(const h16* rowk, int lane) {
  int half = (lane >> 4) & 1;
  v8h lo = *(const v8h*)(rowk + 16 * half);
  v8h hi = *(const v8h*)(rowk + 16 * half + 8);
  v16h a;
#pragma unroll
  for (int e = 0; e < 8; ++e) { a[e] = lo[e]; a[e + 8] = hi[e]; }
  return a;
}

// B fragment: 16 consecutive halves belonging to this lane (K = 16*half + e).
DEV v16h load_b_frag(const h16* p) {
  v8h lo = *(const v8h*)p;
  v8h hi = *(const v8h*)(p + 8);
  v16h b;
#pragma unroll
  for (int e = 0; e < 8; ++e) { b[e] = lo[e]; b[e + 8] = hi[e]; }
  return b;
}

DEV float sigf(float x) { return 1.0f / (1.0f + __expf(-x)); }

// ---------------------------------------------------------------------------
// Generic WMMA GEMM: C[M,Nd] = A[M,K](f16,row-major) * Bp(f16, fragment-packed)
// block = 8 waves; wave tile 32(M) x 64(N) -> 8 WMMAs per k-step with B-frag
// reuse across both M rows; block tile 128 x 128. Double-buffered operand
// fetch so WMMA issue overlaps the next k-step's global_load_b128s.
// Batched via grid.z; generic C strides allow the channel-scattered output of
// the triangle einsum.
// ---------------------------------------------------------------------------
__global__ __launch_bounds__(256) void gemm_kernel(
    const h16* __restrict__ A, const h16* __restrict__ Bp, void* __restrict__ C,
    int M, int K, int Nd,
    long aBatch, long bBatch, long cBatch,
    long cRow, long cCol, int outF16)
{
  int lane = threadIdx.x & 31;
  int wave = threadIdx.x >> 5;
  int wm = wave >> 1, wn = wave & 1;
  int m0 = blockIdx.x * 128 + wm * 32;
  int ntBase = blockIdx.y * 8 + wn * 4;
  int nl = lane & 15, half = lane >> 4;
  const h16* Ab = A + (long)blockIdx.z * aBatch;
  const h16* Bb = Bp + (long)blockIdx.z * bBatch + (long)lane * 16;
  v8f acc[8] = {};
  int ktiles = K >> 5;
  long ntStride = (long)(Nd >> 4) * 512;
  const h16* arow0 = Ab + (long)(m0 + nl) * K;
  const h16* arow1 = arow0 + (long)16 * K;
  const h16* bcol = Bb + (long)ntBase * 512;

  v16h a0 = load_a_frag(arow0, lane);
  v16h a1 = load_a_frag(arow1, lane);
  v16h b0 = load_b_frag(bcol);
  v16h b1 = load_b_frag(bcol + 512);
  v16h b2 = load_b_frag(bcol + 1024);
  v16h b3 = load_b_frag(bcol + 1536);

  for (int kt = 0; kt < ktiles; ++kt) {
    int kn = (kt + 1 < ktiles) ? (kt + 1) : kt;
    // prefetch next k-step's fragments (dead-last-iteration redundancy is
    // cheaper than a branch); compiler waits on partial loadcnt only.
    v16h na0 = load_a_frag(arow0 + kn * 32, lane);
    v16h na1 = load_a_frag(arow1 + kn * 32, lane);
    const h16* bb = bcol + (long)kn * ntStride;
    v16h nb0 = load_b_frag(bb);
    v16h nb1 = load_b_frag(bb + 512);
    v16h nb2 = load_b_frag(bb + 1024);
    v16h nb3 = load_b_frag(bb + 1536);
    acc[0] = wmma_f16(a0, b0, acc[0]);
    acc[4] = wmma_f16(a1, b0, acc[4]);
    acc[1] = wmma_f16(a0, b1, acc[1]);
    acc[5] = wmma_f16(a1, b1, acc[5]);
    acc[2] = wmma_f16(a0, b2, acc[2]);
    acc[6] = wmma_f16(a1, b2, acc[6]);
    acc[3] = wmma_f16(a0, b3, acc[3]);
    acc[7] = wmma_f16(a1, b3, acc[7]);
    a0 = na0; a1 = na1; b0 = nb0; b1 = nb1; b2 = nb2; b3 = nb3;
  }
  long cz = (long)blockIdx.z * cBatch;
#pragma unroll
  for (int r2 = 0; r2 < 2; ++r2) {
#pragma unroll
    for (int t = 0; t < 4; ++t) {
      int ncol = (ntBase + t) * 16 + nl;
#pragma unroll
      for (int r = 0; r < 8; ++r) {
        int m = m0 + r2 * 16 + r + 8 * half;
        long addr = cz + (long)m * cRow + (long)ncol * cCol;
        if (outF16) ((h16*)C)[addr] = (h16)acc[r2 * 4 + t][r];
        else        ((float*)C)[addr] = acc[r2 * 4 + t][r];
      }
    }
  }
}

// f32 weight [K,Nd] -> f16 B-fragment packing: tile(kt,nt) holds 512 halves,
// lane = 16*half + (n&15), elem e = k&15.
__global__ __launch_bounds__(256) void repack_w_kernel(
    const float* __restrict__ W, h16* __restrict__ Bp, int K, int Nd)
{
  int t = blockIdx.x * 256 + threadIdx.x;
  int k = t / Nd, n = t % Nd;
  int kt = k >> 5, rr = k & 31, hh = rr >> 4, e = rr & 15;
  int nt = n >> 4, nn = n & 15;
  Bp[((long)(kt * (Nd >> 4) + nt) << 9) + ((hh * 16 + nn) << 4) + e] = (h16)W[t];
}

// LayerNorm over last dim C (<=384), one wave per row, emit f16.
// trans=1: read source row transposed within a 256x256 grid of rows.
__global__ __launch_bounds__(256) void ln_kernel(
    const float* __restrict__ x, const float* __restrict__ g,
    const float* __restrict__ b, h16* __restrict__ out,
    int rows, int C, int trans)
{
  int lane = threadIdx.x & 31;
  int row = blockIdx.x * 8 + (threadIdx.x >> 5);
  if (row >= rows) return;
  long src = trans ? ((long)(row & 255) * 256 + (row >> 8)) : (long)row;
  const float* xp = x + src * C;
  int cnt = C >> 5;
  float v[12];
  float s = 0.f;
  for (int i = 0; i < cnt; ++i) { v[i] = xp[lane + i * 32]; s += v[i]; }
  for (int o = 16; o; o >>= 1) s += __shfl_xor(s, o, 32);
  float mean = s / (float)C;
  float q = 0.f;
  for (int i = 0; i < cnt; ++i) { float d = v[i] - mean; q += d * d; }
  for (int o = 16; o; o >>= 1) q += __shfl_xor(q, o, 32);
  float rstd = rsqrtf(q / (float)C + 1e-5f);
  h16* op = out + (long)row * C;
  for (int i = 0; i < cnt; ++i) {
    int c = lane + i * 32;
    op[c] = (h16)((v[i] - mean) * rstd * g[c] + b[c]);
  }
}

// a/b gate combine for triangle multiply; writes A-row-major [c][f][s] or
// B-fragment layout (K=f, Ncol=s) per 256x256 channel plane.
__global__ __launch_bounds__(256) void combine_sigmul_kernel(
    const float* __restrict__ G, const float* __restrict__ P,
    h16* __restrict__ out, int bfrag, int swap)
{
  long t = (long)blockIdx.x * 256 + threadIdx.x;
  int c = (int)(t & 127);
  long mn = t >> 7;
  int n = (int)(mn & 255), m = (int)(mn >> 8);
  float val = sigf(G[t]) * P[t];
  int f = swap ? n : m, s2 = swap ? m : n;
  long cb = (long)c << 16;
  if (!bfrag) {
    out[cb + (long)f * 256 + s2] = (h16)val;
  } else {
    int kt = f >> 5, rr = f & 31, hh = rr >> 4, e = rr & 15;
    int nt = s2 >> 4, nn = s2 & 15;
    out[cb + ((long)(kt * 16 + nt) << 9) + ((hh * 16 + nn) << 4) + e] = (h16)val;
  }
}

// pair-bias projection: bias[h][row][col] = sum_c zl[row*256+col][c]*wb[c][h]
// (+ optional mask term). H = 4 (tri) or 16 (single attention).
__global__ __launch_bounds__(256) void bias_kernel(
    const h16* __restrict__ zl, const float* __restrict__ wb,
    const float* __restrict__ mask, float* __restrict__ bias, int H)
{
  int t = blockIdx.x * 256 + threadIdx.x;
  const h16* zp = zl + (long)t * 128;
  float acc[16];
  for (int h = 0; h < H; ++h) acc[h] = 0.f;
  for (int c = 0; c < 128; ++c) {
    float z = (float)zp[c];
    for (int h = 0; h < H; ++h) acc[h] += z * wb[c * H + h];
  }
  float mterm = mask ? (mask[t] - 1.0f) * 1e9f : 0.f;
  int j = t >> 8, k = t & 255;
  for (int h = 0; h < H; ++h)
    bias[(((long)h * 256 + j) << 8) + k] = acc[h] + mterm;
}

// V transpose: vt[p][d][k] = (f16) v[(rowBase(p)+k)*ld + colOff(p)+d]
__global__ __launch_bounds__(256) void vtrans_kernel(
    const float* __restrict__ v, h16* __restrict__ vt, int ld, int mode, int total)
{
  int t = blockIdx.x * 256 + threadIdx.x;
  if (t >= total) return;
  int k = t & 255;
  int d = (t >> 8) & 31;
  int p = t >> 13;
  int rowBase, colOff;
  if (mode == 0) { rowBase = (p & 255) * 256; colOff = (p >> 8) * 32; }
  else           { rowBase = 0;               colOff = p * 32; }
  vt[t] = (h16)v[(long)(rowBase + k) * ld + colOff + d];
}

// One wave computes 16 query rows vs 256 keys, d=32: QK^T WMMAs -> LDS
// softmax (shuffle reductions) -> f16 P panel -> PV WMMAs.
// mode 0: tri (p = h*256+i, ld=128); mode 1: single attn (p = head, ld=512).
__global__ __launch_bounds__(32) void attn_kernel(
    const h16* __restrict__ q16, const h16* __restrict__ k16,
    const h16* __restrict__ vt, const float* __restrict__ bias,
    float* __restrict__ obuf, int ld, int mode, float scale)
{
  __shared__ float sS[16 * 256];
  __shared__ h16   sP[16 * 256];
  int lane = threadIdx.x;
  int nl = lane & 15, half = lane >> 4;
  int jt = blockIdx.x, p = blockIdx.y;
  int rowBase, colOff, hb;
  if (mode == 0) { rowBase = (p & 255) * 256; colOff = (p >> 8) * 32; hb = p >> 8; }
  else           { rowBase = 0;               colOff = p * 32;        hb = p; }
  const float* bb = bias + ((long)(hb * 256 + jt * 16)) * 256;
  v16h aq = load_a_frag(q16 + (long)(rowBase + jt * 16 + nl) * ld + colOff, lane);
  for (int kt = 0; kt < 16; ++kt) {
    v16h bk = load_b_frag(k16 + (long)(rowBase + kt * 16 + nl) * ld + colOff + 16 * half);
    v8f acc = {};
    acc = wmma_f16(aq, bk, acc);
#pragma unroll
    for (int r = 0; r < 8; ++r) {
      int sr = r + 8 * half, sc = kt * 16 + nl;
      sS[sr * 256 + sc] = acc[r] * scale + bb[sr * 256 + sc];
    }
  }
  __syncthreads();
  for (int r = 0; r < 16; ++r) {
    float* rp = sS + r * 256 + lane * 8;
    float vv[8];
    float mx = -1e30f;
#pragma unroll
    for (int e = 0; e < 8; ++e) { vv[e] = rp[e]; mx = fmaxf(mx, vv[e]); }
    for (int o = 16; o; o >>= 1) mx = fmaxf(mx, __shfl_xor(mx, o, 32));
    float sm = 0.f;
#pragma unroll
    for (int e = 0; e < 8; ++e) { vv[e] = __expf(vv[e] - mx); sm += vv[e]; }
    for (int o = 16; o; o >>= 1) sm += __shfl_xor(sm, o, 32);
    float inv = 1.f / sm;
    h16* pp = sP + r * 256 + lane * 8;
#pragma unroll
    for (int e = 0; e < 8; ++e) pp[e] = (h16)(vv[e] * inv);
  }
  __syncthreads();
  v8f o0 = {}, o1 = {};
  for (int kc = 0; kc < 8; ++kc) {
    v16h ap = load_a_frag(sP + nl * 256 + kc * 32, lane);
    v16h b0 = load_b_frag(vt + ((long)p * 32 + nl) * 256 + kc * 32 + 16 * half);
    v16h b1 = load_b_frag(vt + ((long)p * 32 + 16 + nl) * 256 + kc * 32 + 16 * half);
    o0 = wmma_f16(ap, b0, o0);
    o1 = wmma_f16(ap, b1, o1);
  }
#pragma unroll
  for (int r = 0; r < 8; ++r) {
    int m = jt * 16 + r + 8 * half;
    long base = (long)(rowBase + m) * ld + colOff;
    obuf[base + nl]      = o0[r];
    obuf[base + 16 + nl] = o1[r];
  }
}

// --------------------------- elementwise kernels ---------------------------
__global__ __launch_bounds__(256) void combine_go_kernel(
    const float* __restrict__ g, const float* __restrict__ o,
    h16* __restrict__ out, int n)
{
  int t = blockIdx.x * 256 + threadIdx.x;
  if (t < n) out[t] = (h16)(sigf(g[t]) * o[t]);
}

__global__ __launch_bounds__(256) void resid_gate_kernel(
    float* __restrict__ dst, const float* __restrict__ g,
    const float* __restrict__ x, int n)
{
  int t = blockIdx.x * 256 + threadIdx.x;
  if (t < n) dst[t] += sigf(g[t]) * x[t];
}

__global__ __launch_bounds__(256) void resid_add_kernel(
    float* __restrict__ dst, const float* __restrict__ src, int n, int trans)
{
  int t = blockIdx.x * 256 + threadIdx.x;
  if (t >= n) return;
  if (!trans) { dst[t] += src[t]; return; }
  int c = t & 127;
  int ij = t >> 7;
  int i = ij >> 8, j = ij & 255;
  dst[(((long)j * 256 + i) << 7) + c] += src[t];
}

__global__ __launch_bounds__(256) void silu_combine_kernel(
    const float* __restrict__ a, const float* __restrict__ b,
    h16* __restrict__ out, int n)
{
  int t = blockIdx.x * 256 + threadIdx.x;
  if (t < n) { float x = a[t]; out[t] = (h16)(x * sigf(x) * b[t]); }
}

__global__ __launch_bounds__(256) void addbq_kernel(
    h16* __restrict__ q, const float* __restrict__ bq, int n)
{
  int t = blockIdx.x * 256 + threadIdx.x;
  if (t < n) q[t] = (h16)((float)q[t] + bq[t & 511]);
}

// ---------------------------------------------------------------------------
extern "C" void kernel_launch(void* const* d_in, const int* in_sizes, int n_in,
                              void* d_out, int out_size, void* d_ws, size_t ws_size,
                              hipStream_t stream)
{
  (void)in_sizes; (void)n_in; (void)out_size; (void)ws_size;
  const size_t MB = 1048576;
  char* ws = (char*)d_ws;
  float* F32A = (float*)(ws + 0 * MB);
  float* F32B = (float*)(ws + 32 * MB);
  float* F32C = (float*)(ws + 64 * MB);
  float* F32G = (float*)(ws + 96 * MB);
  h16* ZL  = (h16*)(ws + 128 * MB);
  h16* T1  = (h16*)(ws + 144 * MB);
  h16* T2  = (h16*)(ws + 160 * MB);
  h16* T3  = (h16*)(ws + 176 * MB);
  h16* H16 = (h16*)(ws + 192 * MB);
  float* BIAS = (float*)(ws + 200 * MB);
  char* WP = ws + 204 * MB;
  char* SMALL = ws + 220 * MB;
  h16* s16   = (h16*)(SMALL);
  h16* q16s  = (h16*)(SMALL + 256 * 1024);
  h16* k16s  = (h16*)(SMALL + 512 * 1024);
  h16* vts   = (h16*)(SMALL + 768 * 1024);
  h16* go16s = (h16*)(SMALL + 1024 * 1024);

  auto P = [&](int i) { return (const float*)d_in[i]; };

  float* outSingle = (float*)d_out;
  float* outPair   = outSingle + 256 * 384;
  hipMemcpyAsync(outSingle, d_in[0], (size_t)256 * 384 * 4, hipMemcpyDeviceToDevice, stream);
  hipMemcpyAsync(outPair,   d_in[1], (size_t)65536 * 128 * 4, hipMemcpyDeviceToDevice, stream);

  size_t wp = 0;
  auto repack = [&](int idx, int K, int Nd) {
    h16* dst = (h16*)(WP + wp);
    wp += (size_t)K * Nd * 2;
    repack_w_kernel<<<(K * Nd) / 256, 256, 0, stream>>>(P(idx), dst, K, Nd);
    return dst;
  };
  auto gemm = [&](const h16* A, const h16* Bp, void* C, int M, int K, int Nd, bool f16out) {
    dim3 g(M / 128, Nd / 128, 1);
    gemm_kernel<<<g, 256, 0, stream>>>(A, Bp, C, M, K, Nd,
                                       0, 0, 0, (long)Nd, 1, f16out ? 1 : 0);
  };

  // ---- weight repacking (f32 -> WMMA B-fragment f16) ----
  h16 *tmoAG = repack(5, 128, 128),  *tmoAP = repack(6, 128, 128);
  h16 *tmoBG = repack(7, 128, 128),  *tmoBP = repack(8, 128, 128);
  h16 *tmoOG = repack(9, 128, 128),  *tmoOP = repack(12, 128, 128);
  h16 *tmiAG = repack(15, 128, 128), *tmiAP = repack(16, 128, 128);
  h16 *tmiBG = repack(17, 128, 128), *tmiBP = repack(18, 128, 128);
  h16 *tmiOG = repack(19, 128, 128), *tmiOP = repack(22, 128, 128);
  h16 *tasWQ = repack(25, 128, 128), *tasWK = repack(26, 128, 128);
  h16 *tasWV = repack(27, 128, 128), *tasWG = repack(29, 128, 128);
  h16 *tasWO = repack(30, 128, 128);
  h16 *taeWQ = repack(33, 128, 128), *taeWK = repack(34, 128, 128);
  h16 *taeWV = repack(35, 128, 128), *taeWG = repack(37, 128, 128);
  h16 *taeWO = repack(38, 128, 128);
  h16 *ptW1 = repack(41, 128, 512), *ptW2 = repack(42, 128, 512), *ptW3 = repack(43, 512, 128);
  h16 *saWQ = repack(48, 384, 512), *saWK = repack(50, 384, 512);
  h16 *saWV = repack(51, 384, 512), *saWG = repack(52, 384, 512);
  h16 *saWO = repack(54, 512, 384);
  h16 *stW1 = repack(57, 384, 1536), *stW2 = repack(58, 384, 1536), *stW3 = repack(59, 1536, 384);

  const int MPAIR = 65536;
  const float SCALE = 0.17677669529663689f; // 1/sqrt(32)

  // ---- triangle multiplicative update ----
  auto tri_mult = [&](int lnG, int lnB, h16* ag, h16* ap, h16* bg, h16* bp,
                      h16* og, int olnG, int olnB, h16* op, bool outgoing) {
    ln_kernel<<<8192, 256, 0, stream>>>(outPair, P(lnG), P(lnB), ZL, MPAIR, 128, 0);
    gemm(ZL, ag, F32A, MPAIR, 128, 128, false);
    gemm(ZL, ap, F32B, MPAIR, 128, 128, false);
    combine_sigmul_kernel<<<32768, 256, 0, stream>>>(F32A, F32B, T1, 0, outgoing ? 0 : 1);
    gemm(ZL, bg, F32A, MPAIR, 128, 128, false);
    gemm(ZL, bp, F32B, MPAIR, 128, 128, false);
    combine_sigmul_kernel<<<32768, 256, 0, stream>>>(F32A, F32B, T2, 1, outgoing ? 1 : 0);
    gemm(ZL, og, F32G, MPAIR, 128, 128, false);
    // per-channel 256x256x256 batched GEMM -> t[i,j,c] (scatter stride 128)
    dim3 bgrid(2, 2, 128);
    gemm_kernel<<<bgrid, 256, 0, stream>>>(T1, T2, F32A, 256, 256, 256,
                                           65536, 65536, 1, 256L * 128, 128, 0);
    ln_kernel<<<8192, 256, 0, stream>>>(F32A, P(olnG), P(olnB), T3, MPAIR, 128, 0);
    gemm(T3, op, F32B, MPAIR, 128, 128, false);
    resid_gate_kernel<<<32768, 256, 0, stream>>>(outPair, F32G, F32B, 8388608);
  };

  // ---- triangle attention ----
  auto tri_attn = [&](int lnG, int lnB, h16* wq, h16* wk, h16* wv, int wbIdx,
                      h16* wg, h16* wo, bool ending) {
    ln_kernel<<<8192, 256, 0, stream>>>(outPair, P(lnG), P(lnB), ZL, MPAIR, 128, ending ? 1 : 0);
    gemm(ZL, wq, T1, MPAIR, 128, 128, true);
    gemm(ZL, wk, T2, MPAIR, 128, 128, true);
    gemm(ZL, wv, F32A, MPAIR, 128, 128, false);
    vtrans_kernel<<<32768, 256, 0, stream>>>(F32A, T3, 128, 0, 8388608);
    gemm(ZL, wg, F32G, MPAIR, 128, 128, false);
    bias_kernel<<<256, 256, 0, stream>>>(ZL, P(wbIdx), nullptr, BIAS, 4);
    attn_kernel<<<dim3(16, 1024), 32, 0, stream>>>(T1, T2, T3, BIAS, F32A, 128, 0, SCALE);
    combine_go_kernel<<<32768, 256, 0, stream>>>(F32G, F32A, T1, 8388608);
    gemm(T1, wo, F32B, MPAIR, 128, 128, false);
    resid_add_kernel<<<32768, 256, 0, stream>>>(outPair, F32B, 8388608, ending ? 1 : 0);
  };

  // ---- SwiGLU transition (chunked) ----
  auto transition = [&](float* x, int rows, int C, int lnG, int lnB,
                        h16* w1, h16* w2, h16* w3, int chunk) {
    ln_kernel<<<(rows + 7) / 8, 256, 0, stream>>>(x, P(lnG), P(lnB), ZL, rows, C, 0);
    int H = 4 * C;
    for (int r0 = 0; r0 < rows; r0 += chunk) {
      const h16* Ac = ZL + (long)r0 * C;
      gemm(Ac, w1, F32A, chunk, C, H, false);
      gemm(Ac, w2, F32B, chunk, C, H, false);
      silu_combine_kernel<<<(chunk * H) / 256, 256, 0, stream>>>(F32A, F32B, H16, chunk * H);
      gemm(H16, w3, F32C, chunk, H, C, false);
      resid_add_kernel<<<(chunk * C) / 256, 256, 0, stream>>>(x + (long)r0 * C, F32C, chunk * C, 0);
    }
  };

  // ================= block =================
  tri_mult(3, 4, tmoAG, tmoAP, tmoBG, tmoBP, tmoOG, 10, 11, tmoOP, true);
  tri_mult(13, 14, tmiAG, tmiAP, tmiBG, tmiBP, tmiOG, 20, 21, tmiOP, false);
  tri_attn(23, 24, tasWQ, tasWK, tasWV, 28, tasWG, tasWO, false);
  tri_attn(31, 32, taeWQ, taeWK, taeWV, 36, taeWG, taeWO, true);
  transition(outPair, MPAIR, 128, 39, 40, ptW1, ptW2, ptW3, 8192);

  // ---- single attention with pair bias ----
  ln_kernel<<<32, 256, 0, stream>>>(outSingle, P(44), P(45), s16, 256, 384, 0);
  gemm(s16, saWQ, q16s, 256, 384, 512, true);
  addbq_kernel<<<512, 256, 0, stream>>>(q16s, P(49), 131072);
  gemm(s16, saWK, k16s, 256, 384, 512, true);
  gemm(s16, saWV, F32A, 256, 384, 512, false);
  vtrans_kernel<<<512, 256, 0, stream>>>(F32A, vts, 512, 1, 131072);
  gemm(s16, saWG, F32B, 256, 384, 512, false);
  ln_kernel<<<8192, 256, 0, stream>>>(outPair, P(46), P(47), ZL, MPAIR, 128, 0);
  bias_kernel<<<256, 256, 0, stream>>>(ZL, P(53), (const float*)d_in[2], BIAS, 16);
  attn_kernel<<<dim3(16, 16), 32, 0, stream>>>(q16s, k16s, vts, BIAS, F32C, 512, 1, SCALE);
  combine_go_kernel<<<512, 256, 0, stream>>>(F32B, F32C, go16s, 131072);
  gemm(go16s, saWO, F32A, 256, 512, 384, false);
  resid_add_kernel<<<384, 256, 0, stream>>>(outSingle, F32A, 98304, 0);

  transition(outSingle, 256, 384, 55, 56, stW1, stW2, stW3, 256);
}
